// GTN_39573828665768
// MI455X (gfx1250) — compile-verified, matrix operand order checked by reading, not run on hardware
//
#include <hip/hip_runtime.h>
#include <hip/hip_bf16.h>

// Problem constants (match reference)
#define NN      100000
#define EE      600000
#define IN_DIM  64
#define HH      128
#define LL      3
#define GG      256
#define CC      10
#define QK      512       // 4*H: [Q | K | V | S] fused GEMM output width
#define EPS_BN  1e-5f

typedef __attribute__((ext_vector_type(16))) _Float16 v16h;
typedef __attribute__((ext_vector_type(8)))  _Float16 v8h;
typedef __attribute__((ext_vector_type(8)))  float    v8f;

#if defined(__AMDGCN__) && __has_builtin(__builtin_amdgcn_global_load_async_to_lds_b128) && \
    __has_builtin(__builtin_amdgcn_s_wait_asynccnt)
#define HAS_ASYNC_LDS 1
typedef int v4i __attribute__((vector_size(16)));
typedef __attribute__((address_space(1))) v4i* gv4i_p;   // global int4*
typedef __attribute__((address_space(3))) v4i* lv4i_p;   // LDS int4*
#else
#define HAS_ASYNC_LDS 0
#endif

// ---------------------------------------------------------------- utilities
__global__ void fill_f32(float* __restrict__ p, float v, int n) {
    int i = blockIdx.x * blockDim.x + threadIdx.x;
    if (i < n) p[i] = v;
}

__global__ void cvt_f16(const float* __restrict__ x, _Float16* __restrict__ y, int n) {
    int i = blockIdx.x * blockDim.x + threadIdx.x;
    if (i < n) y[i] = (_Float16)x[i];
}

// Pack [Wq|Wk|Wv|Ws] (each [K x 128] row-major f32) into f16 W^T [512 x K]
// (row j of wt = column j of the fused W), plus fused bias [512].
__global__ void pack_weights(const float* __restrict__ Wq, const float* __restrict__ Wk,
                             const float* __restrict__ Wv, const float* __restrict__ Ws,
                             const float* __restrict__ bq, const float* __restrict__ bk,
                             const float* __restrict__ bv, const float* __restrict__ bs,
                             _Float16* __restrict__ wt, float* __restrict__ bc, int K) {
    int idx = blockIdx.x * blockDim.x + threadIdx.x;
    if (idx >= QK * K) return;
    int j = idx / K;
    int k = idx - j * K;
    const float* W = (j < HH) ? Wq : (j < 2 * HH) ? Wk : (j < 3 * HH) ? Wv : Ws;
    int jj = j & (HH - 1);
    wt[idx] = (_Float16)W[k * HH + jj];
    if (k == 0) {
        const float* B = (j < HH) ? bq : (j < 2 * HH) ? bk : (j < 3 * HH) ? bv : bs;
        bc[j] = B[jj];
    }
}

// ------------------------------------------------------------- WMMA GEMM
// out[N x 512] = A[N x K](f16) @ W^T-rows(f16) + bias, f32 accumulate.
// Block: 8 waves, 256 rows x 64 cols. B strip (64 cols x K <= 16KB) staged in
// LDS via CDNA5 async global->LDS copy; each wave computes a 32x64 tile
// (2 A frags x 4 B frags -> 8 v_wmma per K-step of 32).
// A frag (16x32 f16): lane l: m=l&15, half=l>>4; VGPR0-3 = K(half*8+0..7),
// VGPR4-7 = K(16+half*8+0..7).  B frag mirrors with column n=l&15.
__global__ __launch_bounds__(256) void gemm_qkvs(const _Float16* __restrict__ A,
                                                 const _Float16* __restrict__ WT,
                                                 const float* __restrict__ bc,
                                                 float* __restrict__ out,
                                                 int Nr, int K) {
    __shared__ _Float16 Bs[64 * HH];            // 16 KB max (K <= 128)
    const int n0base = blockIdx.y * 64;

    // ---- stage B strip: 64*K f16, linear copy in 16B chunks -------------
    const _Float16* gsrc = WT + (size_t)n0base * K;
    const int nch = (64 * K) / 8;
    for (int c = threadIdx.x; c < nch; c += 256) {
#if HAS_ASYNC_LDS
        __builtin_amdgcn_global_load_async_to_lds_b128(
            (gv4i_p)(gsrc + c * 8), (lv4i_p)(&Bs[c * 8]), 0, 0);
#else
        *(v8h*)(&Bs[c * 8]) = *(const v8h*)(gsrc + c * 8);
#endif
    }
#if HAS_ASYNC_LDS
    __builtin_amdgcn_s_wait_asynccnt(0);
#endif
    __syncthreads();

    const int lane  = threadIdx.x & 31;
    const int wave  = threadIdx.x >> 5;
    const int mlane = lane & 15;
    const int half  = lane >> 4;
    const int m0    = blockIdx.x * 256 + wave * 32;   // wave covers 32 rows
    if (m0 >= Nr) return;                             // wave-uniform, after barrier

    int r0 = m0 + mlane;      if (r0 >= Nr) r0 = Nr - 1;
    int r1 = m0 + 16 + mlane; if (r1 >= Nr) r1 = Nr - 1;
    const _Float16* ap0 = A + (size_t)r0 * K;
    const _Float16* ap1 = A + (size_t)r1 * K;

    v8f acc[2][4] = {};
    for (int kk = 0; kk < K; kk += 32) {
        v16h a0, a1;
        {
            v8h c0 = *(const v8h*)(ap0 + kk + half * 8);
            v8h c1 = *(const v8h*)(ap0 + kk + 16 + half * 8);
            v8h d0 = *(const v8h*)(ap1 + kk + half * 8);
            v8h d1 = *(const v8h*)(ap1 + kk + 16 + half * 8);
#pragma unroll
            for (int i = 0; i < 8; ++i) {
                a0[i] = c0[i]; a0[i + 8] = c1[i];
                a1[i] = d0[i]; a1[i + 8] = d1[i];
            }
        }
        v16h b[4];
#pragma unroll
        for (int t = 0; t < 4; ++t) {
            const _Float16* bp = &Bs[(size_t)(t * 16 + mlane) * K];
            v8h c0 = *(const v8h*)(bp + kk + half * 8);
            v8h c1 = *(const v8h*)(bp + kk + 16 + half * 8);
#pragma unroll
            for (int i = 0; i < 8; ++i) { b[t][i] = c0[i]; b[t][i + 8] = c1[i]; }
        }
#pragma unroll
        for (int t = 0; t < 4; ++t)
            acc[0][t] = __builtin_amdgcn_wmma_f32_16x16x32_f16(
                false, a0, false, b[t], (short)0, acc[0][t], false, false);
#pragma unroll
        for (int t = 0; t < 4; ++t)
            acc[1][t] = __builtin_amdgcn_wmma_f32_16x16x32_f16(
                false, a1, false, b[t], (short)0, acc[1][t], false, false);
    }
#pragma unroll
    for (int mi = 0; mi < 2; ++mi) {
#pragma unroll
        for (int t = 0; t < 4; ++t) {
            const int col = n0base + t * 16 + mlane;
            const float bias = bc[col];
#pragma unroll
            for (int r = 0; r < 8; ++r) {
                int row = m0 + mi * 16 + r + 8 * half;  // C/D layout: VGPR r -> M=r(+8)
                if (row < Nr) out[(size_t)row * QK + col] = acc[mi][t][r] + bias;
            }
        }
    }
}

// ------------------------------------------------------- edge attention
__device__ inline void atomic_max_f32(float* addr, float v) {
    // Order-preserving bit trick; softmax is shift-invariant so any finite,
    // consistent per-dst value is exact, but this computes the true max.
    if (!signbit(v)) atomicMax((int*)addr, __float_as_int(v));
    else             atomicMin((unsigned int*)addr, __float_as_uint(v));
}

// 8 lanes per edge: a_e = scale * dot(q[dst], k[src]) over H=128
__global__ void edge_logits(const float* __restrict__ qkvs, const int* __restrict__ src,
                            const int* __restrict__ dst, float* __restrict__ ea,
                            float* __restrict__ em, float scale) {
    int gid = blockIdx.x * blockDim.x + threadIdx.x;
    int e = gid >> 3, sub = gid & 7;
    if (e >= EE) return;
    int s = src[e], d = dst[e];
    const float4* q = (const float4*)(qkvs + (size_t)d * QK);        // cols 0..127
    const float4* k = (const float4*)(qkvs + (size_t)s * QK + HH);   // cols 128..255
    float acc = 0.f;
#pragma unroll
    for (int i = 0; i < 4; ++i) {
        float4 a = q[sub + 8 * i], b = k[sub + 8 * i];
        acc += a.x * b.x + a.y * b.y + a.z * b.z + a.w * b.w;
    }
    acc += __shfl_xor(acc, 1, 32);
    acc += __shfl_xor(acc, 2, 32);
    acc += __shfl_xor(acc, 4, 32);
    if (sub == 0) {
        float a = acc * scale;
        ea[e] = a;
        atomic_max_f32(&em[d], a);
    }
}

__global__ void edge_exp(const int* __restrict__ dst, float* __restrict__ ea,
                         const float* __restrict__ em, float* __restrict__ es) {
    int e = blockIdx.x * blockDim.x + threadIdx.x;
    if (e >= EE) return;
    int d = dst[e];
    float ex = __expf(ea[e] - em[d]);
    ea[e] = ex;                                  // reuse buffer for exp value
    atomicAdd(&es[d], ex);
}

// wave per edge: agg[dst] += w * v[src], lane owns 4 features
__global__ void edge_agg(const float* __restrict__ qkvs, const int* __restrict__ src,
                         const int* __restrict__ dst, const float* __restrict__ ea,
                         const float* __restrict__ es, float* __restrict__ agg) {
    int gid = blockIdx.x * blockDim.x + threadIdx.x;
    int e = gid >> 5, lane = gid & 31;
    if (e >= EE) return;
    int s = src[e], d = dst[e];
    float w = ea[e] / es[d];
    float4 v = *(const float4*)(qkvs + (size_t)s * QK + 2 * HH + lane * 4);
    float* ag = agg + (size_t)d * HH + lane * 4;
    atomicAdd(ag + 0, v.x * w);
    atomicAdd(ag + 1, v.y * w);
    atomicAdd(ag + 2, v.z * w);
    atomicAdd(ag + 3, v.w * w);
}

// xpre += skip (S columns of qkvs, bias already folded in by the GEMM)
__global__ void combine_skip(float* __restrict__ xpre, const float* __restrict__ qkvs) {
    int i = blockIdx.x * blockDim.x + threadIdx.x;
    if (i >= NN * HH) return;
    int row = i >> 7, f = i & 127;
    xpre[i] += qkvs[(size_t)row * QK + 3 * HH + f];
}

// ----------------------------------------------------------- batch norm
__global__ void bn_stats(const float* __restrict__ x, float* __restrict__ sum,
                         float* __restrict__ sq) {
    int f = threadIdx.x;                         // blockDim.x == 128
    int r0 = blockIdx.x * 256;
    int r1 = min(NN, r0 + 256);
    float s = 0.f, s2 = 0.f;
    for (int r = r0; r < r1; ++r) {
        float v = x[(size_t)r * HH + f];
        s += v; s2 += v * v;
    }
    atomicAdd(&sum[f], s);
    atomicAdd(&sq[f], s2);
}

__global__ void bn_apply(const float* __restrict__ x, const float* __restrict__ sum,
                         const float* __restrict__ sq, const float* __restrict__ g,
                         const float* __restrict__ b, float* __restrict__ hout,
                         _Float16* __restrict__ h16) {
    int i = blockIdx.x * blockDim.x + threadIdx.x;
    if (i >= NN * HH) return;
    int f = i & 127;
    const float invN = 1.f / (float)NN;
    float mu  = sum[f] * invN;
    float var = sq[f] * invN - mu * mu;
    float inv = rsqrtf(var + EPS_BN);
    float y = (x[i] - mu) * inv * g[f] + b[f];
    y = fmaxf(y, 0.f);
    hout[i] = y;                 // f32 activations (final layer feeds pooling)
    h16[i]  = (_Float16)y;       // f16 activations for next layer's WMMA GEMM
}

// --------------------------------------------------------------- pooling
__global__ void pool_sum(const float* __restrict__ h, const int* __restrict__ batch,
                         float* __restrict__ pooled) {
    int i = blockIdx.x * blockDim.x + threadIdx.x;
    if (i >= NN * HH) return;
    int row = i >> 7, f = i & 127;
    atomicAdd(&pooled[(size_t)batch[row] * HH + f], h[i]);
}

__global__ void pool_cnt(const int* __restrict__ batch, float* __restrict__ cnt) {
    int n = blockIdx.x * blockDim.x + threadIdx.x;
    if (n < NN) atomicAdd(&cnt[batch[n]], 1.f);
}

__global__ void head(const float* __restrict__ pooled, const float* __restrict__ cnt,
                     const float* __restrict__ Wl, const float* __restrict__ bl,
                     float* __restrict__ out) {
    int idx = blockIdx.x * blockDim.x + threadIdx.x;
    if (idx >= GG * CC) return;
    int gi = idx / CC, c = idx - gi * CC;
    float invc = 1.f / fmaxf(cnt[gi], 1.f);
    float acc = bl[c];
    for (int f = 0; f < HH; ++f)
        acc += pooled[(size_t)gi * HH + f] * invc * Wl[f * CC + c];
    out[idx] = acc;
}

// ---------------------------------------------------------------- driver
extern "C" void kernel_launch(void* const* d_in, const int* in_sizes, int n_in,
                              void* d_out, int out_size, void* d_ws, size_t ws_size,
                              hipStream_t stream) {
    (void)in_sizes; (void)n_in; (void)out_size; (void)ws_size;
    const float* x    = (const float*)d_in[0];
    const int*   ei   = (const int*)d_in[1];
    const int*   src  = ei;
    const int*   dst  = ei + EE;
    const int*   batch = (const int*)d_in[2];
    const float* linW = (const float*)d_in[23];
    const float* linb = (const float*)d_in[24];
    float* out = (float*)d_out;

    // workspace carve-out (~336 MB)
    char* ws = (char*)d_ws;
    size_t off = 0;
    auto carve = [&](size_t bytes) -> char* {
        char* p = ws + off;
        off = (off + bytes + 255) & ~(size_t)255;
        return p;
    };
    _Float16* h16  = (_Float16*)carve((size_t)NN * HH * 2);
    float* qkvs    = (float*)carve((size_t)NN * QK * 4);
    float* xpre    = (float*)carve((size_t)NN * HH * 4);   // agg, then pre-BN
    float* hout    = (float*)carve((size_t)NN * HH * 4);
    float* ea      = (float*)carve((size_t)EE * 4);
    float* em      = (float*)carve((size_t)NN * 4);
    float* es      = (float*)carve((size_t)NN * 4);
    float* bnsum   = (float*)carve(HH * 4);
    float* bnsq    = (float*)carve(HH * 4);
    _Float16* wt   = (_Float16*)carve((size_t)QK * HH * 2);
    float* bc      = (float*)carve(QK * 4);
    float* pooled  = (float*)carve((size_t)GG * HH * 4);
    float* cnt     = (float*)carve(GG * 4);

    const float scale = 0.08838834764831845f;  // 1/sqrt(H)

    // layer 1 input in f16
    cvt_f16<<<(NN * IN_DIM + 255) / 256, 256, 0, stream>>>(x, h16, NN * IN_DIM);

    for (int l = 0; l < LL + 1; ++l) {
        const float *Wq, *Wk, *Wv, *Ws, *bq, *bk, *bv, *bs, *bg, *bb;
        int K;
        if (l == 0) {
            Wq = (const float*)d_in[3];  bq = (const float*)d_in[4];
            Wk = (const float*)d_in[5];  bk = (const float*)d_in[6];
            Wv = (const float*)d_in[7];  bv = (const float*)d_in[8];
            Ws = (const float*)d_in[9];  bs = (const float*)d_in[10];
            bg = (const float*)d_in[11]; bb = (const float*)d_in[12];
            K = IN_DIM;
        } else {
            size_t i = (size_t)(l - 1);
            Wq = (const float*)d_in[13] + i * HH * HH; bq = (const float*)d_in[14] + i * HH;
            Wk = (const float*)d_in[15] + i * HH * HH; bk = (const float*)d_in[16] + i * HH;
            Wv = (const float*)d_in[17] + i * HH * HH; bv = (const float*)d_in[18] + i * HH;
            Ws = (const float*)d_in[19] + i * HH * HH; bs = (const float*)d_in[20] + i * HH;
            bg = (const float*)d_in[21] + i * HH;      bb = (const float*)d_in[22] + i * HH;
            K = HH;
        }
        pack_weights<<<(QK * K + 255) / 256, 256, 0, stream>>>(
            Wq, Wk, Wv, Ws, bq, bk, bv, bs, wt, bc, K);

        dim3 gg((NN + 255) / 256, QK / 64);
        gemm_qkvs<<<gg, 256, 0, stream>>>(h16, wt, bc, qkvs, NN, K);

        fill_f32<<<(NN + 255) / 256, 256, 0, stream>>>(em, -1e30f, NN);
        fill_f32<<<(NN + 255) / 256, 256, 0, stream>>>(es, 0.f, NN);
        fill_f32<<<(NN * HH + 255) / 256, 256, 0, stream>>>(xpre, 0.f, NN * HH);
        fill_f32<<<1, 256, 0, stream>>>(bnsum, 0.f, HH);
        fill_f32<<<1, 256, 0, stream>>>(bnsq, 0.f, HH);

        edge_logits<<<(EE * 8 + 255) / 256, 256, 0, stream>>>(qkvs, src, dst, ea, em, scale);
        edge_exp<<<(EE + 255) / 256, 256, 0, stream>>>(dst, ea, em, es);
        edge_agg<<<(EE * 32 + 255) / 256, 256, 0, stream>>>(qkvs, src, dst, ea, es, xpre);

        combine_skip<<<(NN * HH + 255) / 256, 256, 0, stream>>>(xpre, qkvs);
        bn_stats<<<(NN + 255) / 256, 128, 0, stream>>>(xpre, bnsum, bnsq);
        bn_apply<<<(NN * HH + 255) / 256, 256, 0, stream>>>(xpre, bnsum, bnsq, bg, bb, hout, h16);
    }

    fill_f32<<<(GG * HH + 255) / 256, 256, 0, stream>>>(pooled, 0.f, GG * HH);
    fill_f32<<<1, 256, 0, stream>>>(cnt, 0.f, GG);
    pool_sum<<<(NN * HH + 255) / 256, 256, 0, stream>>>(hout, batch, pooled);
    pool_cnt<<<(NN + 255) / 256, 256, 0, stream>>>(batch, cnt);
    head<<<(GG * CC + 255) / 256, 256, 0, stream>>>(pooled, cnt, linW, linb, out);
}
